// ScaledDotProductAttention_39152921870887
// MI455X (gfx1250) — matrix operand order repaired and checked by reading, not compiled, not run on hardware
//
#include <hip/hip_runtime.h>
#include <hip/hip_bf16.h>

#define BB 4
#define SS 4096
#define DD 256

typedef __bf16 bf16_t;
typedef __attribute__((ext_vector_type(16))) __bf16 v16bf;
typedef __attribute__((ext_vector_type(8)))  float  v8f;

struct alignas(16) BF8 { bf16_t h[8]; };

// Native conversion: compiler emits packed v_cvt for gfx1250
static __device__ __forceinline__ bf16_t f2bf(float f) { return (bf16_t)f; }

// gfx1250 async global->LDS copy, 16 bytes per lane (tracked by ASYNCcnt)
static __device__ __forceinline__ void async_ld_b128(void* lds, const void* g) {
  unsigned       l  = (unsigned)(size_t)lds;          // low 32 bits = LDS offset
  unsigned long long ga = (unsigned long long)g;
  asm volatile("global_load_async_to_lds_b128 %0, %1, off"
               :: "v"(l), "v"(ga) : "memory");
}

// ---------------------------------------------------------------------------
// Prep 1: x fp32 -> bf16, flat.  8 elements per thread.
// ---------------------------------------------------------------------------
__global__ __launch_bounds__(256) void cvt_x_kernel(
    const float* __restrict__ x, bf16_t* __restrict__ xb)
{
  const size_t i = ((size_t)blockIdx.x * 256 + threadIdx.x) * 8;
  float4 f0 = *(const float4*)(x + i);
  float4 f1 = *(const float4*)(x + i + 4);
  BF8 o;
  o.h[0] = f2bf(f0.x); o.h[1] = f2bf(f0.y); o.h[2] = f2bf(f0.z); o.h[3] = f2bf(f0.w);
  o.h[4] = f2bf(f1.x); o.h[5] = f2bf(f1.y); o.h[6] = f2bf(f1.z); o.h[7] = f2bf(f1.w);
  *(BF8*)(xb + i) = o;
}

// ---------------------------------------------------------------------------
// Prep 2: weight transpose+convert: W[d][e] fp32 -> Wt[e][d] bf16.
// ---------------------------------------------------------------------------
__global__ __launch_bounds__(256) void wt_kernel(
    const float* __restrict__ W, bf16_t* __restrict__ Wt)
{
  const int idx = blockIdx.x * 256 + threadIdx.x;      // 65536 elements
  const int d = idx / DD, e = idx % DD;
  Wt[(size_t)e * DD + d] = f2bf(W[(size_t)d * DD + e]);
}

// ---------------------------------------------------------------------------
// Kernel 1: fused Q/K/V projection (one wave = one 16x16 tile of q,k,v).
// ---------------------------------------------------------------------------
__global__ __launch_bounds__(256) void proj_kernel(
    const bf16_t* __restrict__ xb,
    const bf16_t* __restrict__ Wqt, const float* __restrict__ bq,
    const bf16_t* __restrict__ Wkt, const float* __restrict__ bk,
    const bf16_t* __restrict__ Wvt, const float* __restrict__ bv,
    bf16_t* __restrict__ qb, bf16_t* __restrict__ kb, bf16_t* __restrict__ vt)
{
  const int lane = threadIdx.x & 31;
  const int wave = threadIdx.x >> 5;
  const int tile = blockIdx.x * 8 + wave;       // 16384 tiles total
  const int st = tile >> 4;
  const int et = tile & 15;
  const int row0 = st * 16;
  const int e0 = et * 16;
  const int lm = lane & 15;
  const int lh = lane >> 4;

  v8f aq = {}, ak = {}, av = {};

  const bf16_t* xrow = xb + (size_t)(row0 + lm) * DD + lh * 16;
  const bf16_t* wqr = Wqt + (size_t)(e0 + lm) * DD + lh * 16;
  const bf16_t* wkr = Wkt + (size_t)(e0 + lm) * DD + lh * 16;
  const bf16_t* wvr = Wvt + (size_t)(e0 + lm) * DD + lh * 16;

#pragma unroll
  for (int kc = 0; kc < DD; kc += 32) {
    v16bf a, bqv, bkv, bvv;
    BF8 alo = *(const BF8*)(xrow + kc);
    BF8 ahi = *(const BF8*)(xrow + kc + 8);
    BF8 q0 = *(const BF8*)(wqr + kc), q1 = *(const BF8*)(wqr + kc + 8);
    BF8 k0 = *(const BF8*)(wkr + kc), k1 = *(const BF8*)(wkr + kc + 8);
    BF8 v0 = *(const BF8*)(wvr + kc), v1 = *(const BF8*)(wvr + kc + 8);
#pragma unroll
    for (int i = 0; i < 8; ++i) {
      a[i] = alo.h[i];   a[i + 8] = ahi.h[i];
      bqv[i] = q0.h[i];  bqv[i + 8] = q1.h[i];
      bkv[i] = k0.h[i];  bkv[i + 8] = k1.h[i];
      bvv[i] = v0.h[i];  bvv[i + 8] = v1.h[i];
    }
    aq = __builtin_amdgcn_wmma_f32_16x16x32_bf16(false, a, false, bqv, (short)0, aq, false, false);
    ak = __builtin_amdgcn_wmma_f32_16x16x32_bf16(false, a, false, bkv, (short)0, ak, false, false);
    av = __builtin_amdgcn_wmma_f32_16x16x32_bf16(false, a, false, bvv, (short)0, av, false, false);
  }

  const float biasq = bq[e0 + lm], biask = bk[e0 + lm], biasv = bv[e0 + lm];
  const int m0 = lh * 8;

#pragma unroll
  for (int r = 0; r < 8; ++r) {
    const size_t ro = (size_t)(row0 + m0 + r) * DD + e0 + lm;
    qb[ro] = f2bf((aq[r] + biasq) * 0.0625f);   // fold 1/sqrt(256)
    kb[ro] = f2bf(ak[r] + biask);
  }
  const int b  = row0 / SS;
  const int s0 = row0 % SS;
  BF8 vrow;
#pragma unroll
  for (int r = 0; r < 8; ++r) vrow.h[r] = f2bf(av[r] + biasv);
  *(BF8*)(vt + ((size_t)b * DD + e0 + lm) * SS + s0 + m0) = vrow;
}

// ---------------------------------------------------------------------------
// Kernel 2: flash attention with double-buffered async K/V staging in LDS.
// Block = 8 waves sharing one batch; wave owns a 16-query tile.
// Per 32-key chunk: async-copy next K/V chunk, 16 WMMA (QK^T) from LDS,
// online softmax, 16 WMMA (PV) from LDS.
// ---------------------------------------------------------------------------
#define KSTRIDE 528                 // bytes per key row in LDS (512 + 16 pad)
#define VSTRIDE 80                  // bytes per d row in LDS   (64 + 16 pad)
#define KBUFSZ  (32 * KSTRIDE)      // 16896
#define VBUFSZ  (256 * VSTRIDE)     // 20480
#define VOFF    (2 * KBUFSZ)        // 33792
#define POFF    (VOFF + 2 * VBUFSZ) // 74752
#define SMEMSZ  (POFF + 8 * 16 * 40 * 2)  // 84992

__global__ __launch_bounds__(256) void attn_kernel(
    const bf16_t* __restrict__ qb, const bf16_t* __restrict__ kb,
    const bf16_t* __restrict__ vt, float* __restrict__ out)
{
  __shared__ alignas(16) unsigned char smem[SMEMSZ];
  const int t    = threadIdx.x;
  const int lane = t & 31;
  const int wave = t >> 5;
  const int qt = blockIdx.x * 8 + wave;         // 1024 query tiles
  const int b  = qt / (SS / 16);
  const int q0 = (qt % (SS / 16)) * 16;
  const int lm = lane & 15, lh = lane >> 4;

  const v8f vzero = {};
  const bf16_t* kbase = kb + (size_t)b * SS * DD;
  const bf16_t* vbase = vt + (size_t)b * DD * SS;

  // ---- async stage of one 32-key chunk: 8 x b128 per thread ----------------
  const int krow = t >> 5;          // 0..7  (8 rows per pass, 4 passes)
  const int kseg = t & 31;          // 16B segment within 512B key row
  auto stage = [&](int buf, int kc) {
    unsigned char* kB = smem + buf * KBUFSZ;
    unsigned char* vB = smem + VOFF + buf * VBUFSZ;
#pragma unroll
    for (int j = 0; j < 4; ++j) {
      const int r = krow + j * 8;
      async_ld_b128(kB + r * KSTRIDE + kseg * 16,
                    kbase + (size_t)(kc + r) * DD + kseg * 8);
    }
#pragma unroll
    for (int j = 0; j < 4; ++j)
      async_ld_b128(vB + t * VSTRIDE + j * 16,
                    vbase + (size_t)t * SS + kc + j * 8);
  };

  // Preload Q in WMMA A-layout (8 chunks of K=32 over D=256)
  v16bf qa[8];
  {
    const bf16_t* qrow = qb + ((size_t)b * SS + q0 + lm) * DD + lh * 16;
#pragma unroll
    for (int kk = 0; kk < 8; ++kk) {
      BF8 lo = *(const BF8*)(qrow + kk * 32);
      BF8 hi = *(const BF8*)(qrow + kk * 32 + 8);
#pragma unroll
      for (int i = 0; i < 8; ++i) { qa[kk][i] = lo.h[i]; qa[kk][i + 8] = hi.h[i]; }
    }
  }

  v8f acc[16];
#pragma unroll
  for (int tt = 0; tt < 16; ++tt) acc[tt] = vzero;
  float mrow[8], lrow[8];
#pragma unroll
  for (int r = 0; r < 8; ++r) { mrow[r] = -1.0e30f; lrow[r] = 0.f; }

  bf16_t* pw = (bf16_t*)(smem + POFF) + wave * 16 * 40;

  stage(0, 0);                                  // prologue: fill buffer 0

  const int NITER = SS / 32;                    // 128
  for (int ic = 0; ic < NITER; ++ic) {
    const int cur = ic & 1;
    if (ic + 1 < NITER) {
      stage(cur ^ 1, (ic + 1) * 32);
      asm volatile("s_wait_asynccnt 0x8" ::: "memory");  // current chunk landed
    } else {
      asm volatile("s_wait_asynccnt 0x0" ::: "memory");
    }
    __syncthreads();                            // all waves' copies visible

    const unsigned char* kB = smem + cur * KBUFSZ;
    const unsigned char* vB = smem + VOFF + cur * VBUFSZ;

    // ---- S = Q @ K^T over two 16-key column tiles (from LDS) ---------------
    v8f s0 = vzero, s1 = vzero;
    const unsigned char* kr0 = kB + (size_t)lm * KSTRIDE + lh * 32;
    const unsigned char* kr1 = kr0 + 16 * KSTRIDE;
#pragma unroll
    for (int kk = 0; kk < 8; ++kk) {
      v16bf b0, b1;
      BF8 c0 = *(const BF8*)(kr0 + kk * 64);
      BF8 c1 = *(const BF8*)(kr0 + kk * 64 + 16);
      BF8 d0 = *(const BF8*)(kr1 + kk * 64);
      BF8 d1 = *(const BF8*)(kr1 + kk * 64 + 16);
#pragma unroll
      for (int i = 0; i < 8; ++i) {
        b0[i] = c0.h[i]; b0[i + 8] = c1.h[i];
        b1[i] = d0.h[i]; b1[i + 8] = d1.h[i];
      }
      s0 = __builtin_amdgcn_wmma_f32_16x16x32_bf16(false, qa[kk], false, b0, (short)0, s0, false, false);
      s1 = __builtin_amdgcn_wmma_f32_16x16x32_bf16(false, qa[kk], false, b1, (short)0, s1, false, false);
    }

    // ---- online softmax (per-row stats across 16-lane halves) --------------
    float alpha[8];
#pragma unroll
    for (int r = 0; r < 8; ++r) {
      float v = fmaxf(s0[r], s1[r]);
      v = fmaxf(v, __shfl_xor(v, 1, 16));
      v = fmaxf(v, __shfl_xor(v, 2, 16));
      v = fmaxf(v, __shfl_xor(v, 4, 16));
      v = fmaxf(v, __shfl_xor(v, 8, 16));
      const float nm = fmaxf(mrow[r], v);
      alpha[r] = __expf(mrow[r] - nm);
      mrow[r]  = nm;
      const float p0 = __expf(s0[r] - nm);
      const float p1 = __expf(s1[r] - nm);
      float rs = p0 + p1;
      rs += __shfl_xor(rs, 1, 16);
      rs += __shfl_xor(rs, 2, 16);
      rs += __shfl_xor(rs, 4, 16);
      rs += __shfl_xor(rs, 8, 16);
      lrow[r] = lrow[r] * alpha[r] + rs;
      const int m = r + 8 * lh;                  // C-layout row of this vgpr
      pw[m * 40 + lm]      = f2bf(p0);
      pw[m * 40 + 16 + lm] = f2bf(p1);
    }
#pragma unroll
    for (int tt = 0; tt < 16; ++tt)
#pragma unroll
      for (int r = 0; r < 8; ++r) acc[tt][r] *= alpha[r];

    asm volatile("s_wait_dscnt 0x0" ::: "memory"); // wave-local C->A transpose

    v16bf pa;
    {
      const bf16_t* prow = pw + lm * 40 + lh * 16;
      BF8 lo = *(const BF8*)(prow);
      BF8 hi = *(const BF8*)(prow + 8);
#pragma unroll
      for (int i = 0; i < 8; ++i) { pa[i] = lo.h[i]; pa[i + 8] = hi.h[i]; }
    }

    // ---- context += P @ V (from LDS, V rows are d, cols are keys) ----------
#pragma unroll
    for (int tt = 0; tt < 16; ++tt) {
      v16bf bvv;
      const unsigned char* vr = vB + (size_t)(tt * 16 + lm) * VSTRIDE + lh * 32;
      BF8 c0 = *(const BF8*)(vr);
      BF8 c1 = *(const BF8*)(vr + 16);
#pragma unroll
      for (int i = 0; i < 8; ++i) { bvv[i] = c0.h[i]; bvv[i + 8] = c1.h[i]; }
      acc[tt] = __builtin_amdgcn_wmma_f32_16x16x32_bf16(false, pa, false, bvv, (short)0, acc[tt], false, false);
    }
    __syncthreads();                            // buffer free for re-stage
  }

  // ---- normalize and write out (fp32) --------------------------------------
  float inv[8];
#pragma unroll
  for (int r = 0; r < 8; ++r) inv[r] = 1.0f / lrow[r];
  float* orow = out + ((size_t)b * SS + q0) * DD;
#pragma unroll
  for (int tt = 0; tt < 16; ++tt)
#pragma unroll
    for (int r = 0; r < 8; ++r)
      orow[(size_t)(r + 8 * lh) * DD + tt * 16 + lm] = acc[tt][r] * inv[r];
}

extern "C" void kernel_launch(void* const* d_in, const int* in_sizes, int n_in,
                              void* d_out, int out_size, void* d_ws, size_t ws_size,
                              hipStream_t stream) {
  const float* x  = (const float*)d_in[0];
  const float* Wq = (const float*)d_in[1];
  const float* bq = (const float*)d_in[2];
  const float* Wk = (const float*)d_in[3];
  const float* bk = (const float*)d_in[4];
  const float* Wv = (const float*)d_in[5];
  const float* bv = (const float*)d_in[6];
  float* out = (float*)d_out;

  const size_t BSD = (size_t)BB * SS * DD;      // 4,194,304 elements
  bf16_t* qb  = (bf16_t*)d_ws;
  bf16_t* kb  = qb + BSD;
  bf16_t* vt  = kb + BSD;
  bf16_t* xb  = vt + BSD;
  bf16_t* Wqt = xb + BSD;
  bf16_t* Wkt = Wqt + (size_t)DD * DD;
  bf16_t* Wvt = Wkt + (size_t)DD * DD;

  cvt_x_kernel<<<(int)(BSD / 8 / 256), 256, 0, stream>>>(x, xb);
  wt_kernel<<<DD * DD / 256, 256, 0, stream>>>(Wq, Wqt);
  wt_kernel<<<DD * DD / 256, 256, 0, stream>>>(Wk, Wkt);
  wt_kernel<<<DD * DD / 256, 256, 0, stream>>>(Wv, Wvt);

  proj_kernel<<<2048, 256, 0, stream>>>(xb, Wqt, bq, Wkt, bk, Wvt, bv, qb, kb, vt);
  attn_kernel<<<128, 256, 0, stream>>>(qb, kb, vt, out);
}
